// PatchMixerBlock_83717502533685
// MI455X (gfx1250) — matrix operand order repaired and checked by reading, not compiled, not run on hardware
//
#include <hip/hip_runtime.h>
#include <hip/hip_bf16.h>

typedef __attribute__((ext_vector_type(16))) _Float16 v16h;
typedef __attribute__((ext_vector_type(8)))  _Float16 v8h;
typedef __attribute__((ext_vector_type(8)))  float    v8f;

#define PATCH 16
#define BATCH 64
#define CH 3
#define IMG 384
#define HPW 24            // IMG / PATCH
#define LPATCH 576        // HPW * HPW
#define PS 768            // CH * PATCH * PATCH
#define HID 1536
#define MROWS (BATCH * LPATCH)   // 36864
#define LDT 40            // LDS tile row stride in halves (80B, 16B-aligned, pads banks)
#define TILEH (128 * LDT) // halves per tile buffer

#if __has_builtin(__builtin_amdgcn_global_load_async_to_lds_b128)
#define HAVE_ASYNC 1
#else
#define HAVE_ASYNC 0
#endif

// exact pointee type the builtin wants: 16-byte int vector (vector_size style)
typedef int vsi4 __attribute__((vector_size(16)));
typedef __attribute__((address_space(1))) vsi4* as1v4;
typedef __attribute__((address_space(3))) vsi4* as3v4;

__device__ __forceinline__ void async_wait_all() {
#if HAVE_ASYNC
#if __has_builtin(__builtin_amdgcn_s_wait_asynccnt)
  __builtin_amdgcn_s_wait_asynccnt(0);
#else
  asm volatile("s_wait_asynccnt 0x0" ::: "memory");
#endif
#endif
}

// ---------------- fp32 -> f16 convert (weights, transposed) ----------------
// wt[n][k] = (f16) w[k][n];  w: K x N fp32 row-major -> wt: N x K f16 row-major
__global__ __launch_bounds__(256) void cvt_transpose_f16(const float* __restrict__ w,
                                                         _Float16* __restrict__ wt,
                                                         int K, int N) {
  __shared__ float tile[32][33];
  const int kb = blockIdx.x * 32, nb = blockIdx.y * 32;
  const int tx = threadIdx.x & 31, ty = threadIdx.x >> 5;   // 32 x 8
#pragma unroll
  for (int i = 0; i < 32; i += 8)
    tile[ty + i][tx] = w[(size_t)(kb + ty + i) * N + nb + tx];
  __syncthreads();
#pragma unroll
  for (int i = 0; i < 32; i += 8)
    wt[(size_t)(nb + ty + i) * K + kb + tx] = (_Float16)tile[tx][ty + i];
}

// ---------------- patchify + LayerNorm -> f16 rows ----------------
__global__ __launch_bounds__(256) void patch_ln_kernel(const float* __restrict__ x,
                                                       const float* __restrict__ lnw,
                                                       const float* __restrict__ lnb,
                                                       _Float16* __restrict__ y16) {
  const int r  = blockIdx.x;             // row = (b, hp, wp)
  const int b  = r / LPATCH;
  const int l  = r - b * LPATCH;
  const int hp = l / HPW;
  const int wp = l - hp * HPW;
  const int t  = threadIdx.x;

  float v[3];
  float s = 0.f, ss = 0.f;
#pragma unroll
  for (int j = 0; j < 3; ++j) {
    int p  = t + j * 256;                // p = (c, ph, pw)
    int c  = p >> 8, ph = (p >> 4) & 15, pw = p & 15;
    int idx = ((b * CH + c) * IMG + hp * PATCH + ph) * IMG + wp * PATCH + pw;
    float val = x[idx];
    v[j] = val;
    s += val; ss += val * val;
  }
#pragma unroll
  for (int o = 16; o > 0; o >>= 1) {
    s  += __shfl_down(s,  o, 32);
    ss += __shfl_down(ss, o, 32);
  }
  __shared__ float rs[8], rss[8];
  __shared__ float smu, srstd;
  const int wid = t >> 5, lane = t & 31;
  if (lane == 0) { rs[wid] = s; rss[wid] = ss; }
  __syncthreads();
  if (t == 0) {
    float S = 0.f, SS = 0.f;
#pragma unroll
    for (int i = 0; i < 8; ++i) { S += rs[i]; SS += rss[i]; }
    float mu  = S * (1.0f / PS);
    float var = SS * (1.0f / PS) - mu * mu;
    smu = mu; srstd = rsqrtf(var + 1e-5f);
  }
  __syncthreads();
  const float mu = smu, rstd = srstd;
#pragma unroll
  for (int j = 0; j < 3; ++j) {
    int p = t + j * 256;
    float nv = (v[j] - mu) * rstd * lnw[p] + lnb[p];
    y16[(size_t)r * PS + p] = (_Float16)nv;
  }
}

// ---------------- tile copy: 128 rows x 32 halves, contiguous 16B chunks ----------------
// src row-major with leading dim = K (both A and pre-transposed B share this shape).
__device__ __forceinline__ void tile_copy(const _Float16* __restrict__ src, int K,
                                          int row0, int k0, _Float16* dstLds) {
  const int t = threadIdx.x;
#pragma unroll
  for (int c = 0; c < 2; ++c) {
    int chunk = t + c * 256;             // 0..511
    int row   = chunk >> 2;              // 0..127
    int cc    = (chunk & 3) * 8;         // 0,8,16,24
    const _Float16* g = src + (size_t)(row0 + row) * K + k0 + cc;
    _Float16* l = dstLds + row * LDT + cc;
#if HAVE_ASYNC
    __builtin_amdgcn_global_load_async_to_lds_b128(
        (as1v4)(uintptr_t)g, (as3v4)(uintptr_t)l, 0, 0);
#else
    *(v8h*)l = *(const v8h*)g;
#endif
  }
}

__device__ __forceinline__ v16h make_v16(v8h lo, v8h hi) {
  v16h r;
#pragma unroll
  for (int i = 0; i < 8; ++i) { r[i] = lo[i]; r[i + 8] = hi[i]; }
  return r;
}

// Block tile 128x128, 8 waves arranged 4(m) x 2(n); each wave owns 32x64 (2x4 WMMA tiles).
// A: [M][K] f16, ldA == K. Bt: [N][K] f16 (pre-transposed weights), ldBt == K.
// Double-buffered LDS, async global->LDS copies overlap the WMMAs.
__device__ __forceinline__ void gemm_block(const _Float16* __restrict__ A,
                                           const _Float16* __restrict__ Bt,
                                           int K, int m0, int n0,
                                           _Float16* As, _Float16* Bs,
                                           v8f acc[2][4]) {
  const int t    = threadIdx.x;
  const int wid  = t >> 5;
  const int lane = t & 31;
  const int wm   = (wid & 3) * 32;
  const int wn   = (wid >> 2) * 64;
  const int lr   = lane & 15;
  const int lhi  = lane >> 4;

  int buf = 0;
  tile_copy(A,  K, m0, 0, As);
  tile_copy(Bt, K, n0, 0, Bs);
  async_wait_all();
  __syncthreads();

  for (int k0 = 0; k0 < K; k0 += 32) {
    const int nb = buf ^ 1;
    if (k0 + 32 < K) {                   // prefetch next K-slab into the other buffer
      tile_copy(A,  K, m0, k0 + 32, As + nb * TILEH);
      tile_copy(Bt, K, n0, k0 + 32, Bs + nb * TILEH);
    }

    const _Float16* Ab = As + buf * TILEH;
    const _Float16* Bb = Bs + buf * TILEH;

    // fragments per ISA wave32 VGPR layout
    v16h af[2], bf[4];
#pragma unroll
    for (int mi = 0; mi < 2; ++mi) {
      int row = wm + mi * 16 + lr;
      int kb  = lhi * 8;                 // lane<16: K0-7 & K16-23 ; lane>=16: K8-15 & K24-31
      v8h lo = *(const v8h*)(Ab + row * LDT + kb);
      v8h hi = *(const v8h*)(Ab + row * LDT + kb + 16);
      af[mi] = make_v16(lo, hi);
    }
#pragma unroll
    for (int ni = 0; ni < 4; ++ni) {
      int col = wn + ni * 16 + lr;
      int kb  = lhi * 8;                 // lane<16: K0-7 & K16-23 ; lane>=16: K8-15 & K24-31
      v8h lo = *(const v8h*)(Bb + col * LDT + kb);
      v8h hi = *(const v8h*)(Bb + col * LDT + kb + 16);
      bf[ni] = make_v16(lo, hi);
    }
#pragma unroll
    for (int mi = 0; mi < 2; ++mi)
#pragma unroll
      for (int ni = 0; ni < 4; ++ni)
        acc[mi][ni] = __builtin_amdgcn_wmma_f32_16x16x32_f16(
            false, af[mi], false, bf[ni], (short)0, acc[mi][ni], false, false);

    async_wait_all();                    // next-slab copies landed
    __syncthreads();                     // all waves done reading `buf`
    buf = nb;
  }
}

// ---------------- GEMM1: y16 @ w1 + b1, exact GELU -> h16 ----------------
__global__ __launch_bounds__(256) void gemm1_gelu(const _Float16* __restrict__ A,
                                                  const _Float16* __restrict__ Bt,
                                                  const float* __restrict__ bias,
                                                  _Float16* __restrict__ H) {
  __shared__ _Float16 As[2 * TILEH];
  __shared__ _Float16 Bs[2 * TILEH];
  const int m0 = blockIdx.x * 128;
  const int n0 = blockIdx.y * 128;
  v8f acc[2][4];
  v8f z = {};
#pragma unroll
  for (int i = 0; i < 2; ++i)
#pragma unroll
    for (int j = 0; j < 4; ++j) acc[i][j] = z;

  gemm_block(A, Bt, PS, m0, n0, As, Bs, acc);

  const int lane = threadIdx.x & 31;
  const int wid  = threadIdx.x >> 5;
  const int wm = (wid & 3) * 32, wn = (wid >> 2) * 64;
  const int lr = lane & 15, lhi = lane >> 4;
#pragma unroll
  for (int mi = 0; mi < 2; ++mi)
#pragma unroll
    for (int ni = 0; ni < 4; ++ni) {
      int n    = n0 + wn + ni * 16 + lr;
      float bv = bias[n];
#pragma unroll
      for (int r = 0; r < 8; ++r) {
        int m = m0 + wm + mi * 16 + lhi * 8 + r;   // C/D layout: lanes 16-31 = rows 8-15
        float vv = acc[mi][ni][r] + bv;
        vv = 0.5f * vv * (1.0f + erff(vv * 0.70710678118654752f));  // exact GELU
        H[(size_t)m * HID + n] = (_Float16)vv;
      }
    }
}

// ---------------- GEMM2: h16 @ w2 + b2, fold back + residual -> out ----------------
__global__ __launch_bounds__(256) void gemm2_fold(const _Float16* __restrict__ A,
                                                  const _Float16* __restrict__ Bt,
                                                  const float* __restrict__ bias,
                                                  const float* __restrict__ x,
                                                  float* __restrict__ out) {
  __shared__ _Float16 As[2 * TILEH];
  __shared__ _Float16 Bs[2 * TILEH];
  const int m0 = blockIdx.x * 128;
  const int n0 = blockIdx.y * 128;
  v8f acc[2][4];
  v8f z = {};
#pragma unroll
  for (int i = 0; i < 2; ++i)
#pragma unroll
    for (int j = 0; j < 4; ++j) acc[i][j] = z;

  gemm_block(A, Bt, HID, m0, n0, As, Bs, acc);

  const int lane = threadIdx.x & 31;
  const int wid  = threadIdx.x >> 5;
  const int wm = (wid & 3) * 32, wn = (wid >> 2) * 64;
  const int lr = lane & 15, lhi = lane >> 4;
#pragma unroll
  for (int mi = 0; mi < 2; ++mi)
#pragma unroll
    for (int ni = 0; ni < 4; ++ni) {
      int n    = n0 + wn + ni * 16 + lr;           // n = (c, ph, pw)
      int c    = n >> 8, ph = (n >> 4) & 15, pw = n & 15;
      float bv = bias[n];
#pragma unroll
      for (int r = 0; r < 8; ++r) {
        int m  = m0 + wm + mi * 16 + lhi * 8 + r;  // m = (b, hp, wp)
        int b  = m / LPATCH;
        int l  = m - b * LPATCH;
        int hp = l / HPW;
        int wp = l - hp * HPW;
        size_t idx = ((size_t)(b * CH + c) * IMG + hp * PATCH + ph) * IMG + wp * PATCH + pw;
        out[idx] = x[idx] + acc[mi][ni][r] + bv;
      }
    }
}

// ---------------- launch ----------------
extern "C" void kernel_launch(void* const* d_in, const int* in_sizes, int n_in,
                              void* d_out, int out_size, void* d_ws, size_t ws_size,
                              hipStream_t stream) {
  const float* x   = (const float*)d_in[0];
  const float* lnw = (const float*)d_in[1];
  const float* lnb = (const float*)d_in[2];
  const float* w1  = (const float*)d_in[3];
  const float* b1  = (const float*)d_in[4];
  const float* w2  = (const float*)d_in[5];
  const float* b2  = (const float*)d_in[6];
  float* out = (float*)d_out;

  char* ws = (char*)d_ws;
  size_t off = 0;
  auto alloc = [&](size_t bytes) -> char* {
    char* p = ws + off;
    off = (off + bytes + 255) & ~(size_t)255;
    return p;
  };
  _Float16* w1t = (_Float16*)alloc((size_t)PS * HID * sizeof(_Float16));   // [HID][PS]
  _Float16* w2t = (_Float16*)alloc((size_t)HID * PS * sizeof(_Float16));   // [PS][HID]
  _Float16* y16 = (_Float16*)alloc((size_t)MROWS * PS * sizeof(_Float16));
  _Float16* h16 = (_Float16*)alloc((size_t)MROWS * HID * sizeof(_Float16));

  // transpose+convert weights once (L2-resident afterwards)
  cvt_transpose_f16<<<dim3(PS / 32, HID / 32), 256, 0, stream>>>(w1, w1t, PS, HID);
  cvt_transpose_f16<<<dim3(HID / 32, PS / 32), 256, 0, stream>>>(w2, w2t, HID, PS);
  patch_ln_kernel<<<MROWS, 256, 0, stream>>>(x, lnw, lnb, y16);
  gemm1_gelu<<<dim3(MROWS / 128, HID / 128), 256, 0, stream>>>(y16, w1t, b1, h16);
  gemm2_fold<<<dim3(MROWS / 128, PS / 128), 256, 0, stream>>>(h16, w2t, b2, x, out);
}
